// AttributeRetriever_65137474011221
// MI455X (gfx1250) — matrix operand order repaired and checked by reading, not compiled, not run on hardware
//
#include <hip/hip_runtime.h>
#include <hip/hip_bf16.h>
#include <stdint.h>

typedef __attribute__((ext_vector_type(16))) __bf16 v16bf;
typedef __attribute__((ext_vector_type(8)))  float  v8f;
typedef unsigned int v4u  __attribute__((ext_vector_type(4)));
typedef int          v8i_ __attribute__((ext_vector_type(8)));
typedef int          v4i_ __attribute__((ext_vector_type(4)));

#define FEAT     512
#define KSTEPS   16       // 512 / 32
#define TOPK     5
#define NCHUNKS  16       // N-dimension split across blocks
#define NEG_INF  (-__builtin_inff())

#define BROW     1040                 // padded LDS row stride: 1024 B + 16 B pad
#define TROWS    32                   // attr rows per staged (TDM) tile
#define BUFB     (TROWS * BROW)       // one B tile buffer: 33280 B
#define LISTOFF  (2 * BUFB)           // lists start after double buffer (66560)
#define LDS_SZ   (LISTOFF + 256 * 8 * TOPK * 8)   // + 81920 -> 148480 B

union FragBF { uint4 q[2]; v16bf v; };   // 32 bytes = 16 bf16 = one WMMA operand

// ---- round-to-nearest-even f32 -> bf16 (bit pattern), storage is ushort ----
static __device__ __forceinline__ unsigned short f2bf(float f) {
    unsigned int u = __float_as_uint(f);
    unsigned int r = u + 0x7FFFu + ((u >> 16) & 1u);
    return (unsigned short)(r >> 16);
}

// =====================================================================
// TDM: DMA one 32-row x 512-elem bf16 tile (row-major, stride 512) into
// LDS at ldsByteAddr, inserting a 16 B pad after every 1024 B row so the
// compute waves' ds_load_b128 fragment reads are bank-conflict-free.
// D# per CDNA5 ISA 8.3/8.4. Issued by one wave; tracked with TENSORcnt.
// =====================================================================
static __device__ __forceinline__ void tdm_load_tile(const unsigned short* gsrc,
                                                     unsigned ldsByteAddr)
{
    unsigned long long ga = (unsigned long long)(uintptr_t)gsrc;
    v4u g0;
    g0.x = 1u;                                            // count=1 (valid D#)
    g0.y = ldsByteAddr;                                   // lds_addr
    g0.z = (unsigned)(ga & 0xFFFFFFFFu);                  // global_addr[31:0]
    g0.w = (unsigned)((ga >> 32) & 0x1FFFFFFu)            // global_addr[56:32]
         | (2u << 30);                                    // type=2 ("image")
    v8i_ g1;
    g1[0] = (1 << 16)        // data_size = 2 bytes
          | (1 << 20)        // pad_enable
          | (7 << 22)        // pad_interval: every 256 DWORDs (= one 1024 B row)
          | (3 << 25);       // pad_amount: 4 DWORDs (16 B)
    g1[1] = (int)(512u  << 16);  // tensor_dim0 = 512 (elements)
    g1[2] = (int)(32u   << 16);  // tensor_dim1 = 32
    g1[3] = (int)(512u  << 16);  // tile_dim0   = 512
    g1[4] = TROWS;               // tile_dim1 = 32, tile_dim2 = 0
    g1[5] = 512;                 // tensor_dim0_stride = 512 elements
    g1[6] = 0;
    g1[7] = 0;
    v4i_ z4 = {0, 0, 0, 0};
#if defined(__clang_major__) && (__clang_major__ >= 23)
    v8i_ z8 = {0, 0, 0, 0, 0, 0, 0, 0};
    __builtin_amdgcn_tensor_load_to_lds(g0, g1, z4, z4, z8, 0);
#else
    __builtin_amdgcn_tensor_load_to_lds(g0, g1, z4, z4, 0);
#endif
}

// =====================================================================
// K1: per-attr-row L2 norm; write inv_norm (f32) and normalized bf16 row.
// One wave per row: 32 lanes x 16 floats.
// =====================================================================
__global__ void __launch_bounds__(256)
attr_prep_kernel(const float* __restrict__ attr,
                 unsigned short* __restrict__ attrB,
                 float* __restrict__ invn)
{
    const int wave = threadIdx.x >> 5;
    const int lane = threadIdx.x & 31;
    const int row  = blockIdx.x * 8 + wave;

    const float4* src = (const float4*)(attr + (size_t)row * FEAT);
    float4 x[4];
    float ss = 0.f;
    #pragma unroll
    for (int u = 0; u < 4; ++u) {
        x[u] = src[lane * 4 + u];
        ss += x[u].x * x[u].x + x[u].y * x[u].y + x[u].z * x[u].z + x[u].w * x[u].w;
    }
    #pragma unroll
    for (int d = 16; d >= 1; d >>= 1) ss += __shfl_xor(ss, d, 32);
    const float inv = 1.0f / sqrtf(ss);
    if (lane == 0) invn[row] = inv;

    union { unsigned short h[8]; uint4 q; } pk[2];
    #pragma unroll
    for (int u = 0; u < 4; ++u) {
        pk[u >> 1].h[(u & 1) * 4 + 0] = f2bf(x[u].x * inv);
        pk[u >> 1].h[(u & 1) * 4 + 1] = f2bf(x[u].y * inv);
        pk[u >> 1].h[(u & 1) * 4 + 2] = f2bf(x[u].z * inv);
        pk[u >> 1].h[(u & 1) * 4 + 3] = f2bf(x[u].w * inv);
    }
    uint4* dst = (uint4*)(attrB + (size_t)row * FEAT + lane * 16);
    dst[0] = pk[0].q;
    dst[1] = pk[1].q;
}

// =====================================================================
// K2: image f32 -> bf16 (no normalization in reference)
// =====================================================================
__global__ void __launch_bounds__(256)
img_prep_kernel(const float* __restrict__ img, unsigned short* __restrict__ imgB)
{
    const int i = (blockIdx.x * 256 + threadIdx.x) * 4;
    float4 x = *(const float4*)(img + i);
    union { unsigned short h[4]; uint2 q; } pk;
    pk.h[0] = f2bf(x.x); pk.h[1] = f2bf(x.y);
    pk.h[2] = f2bf(x.z); pk.h[3] = f2bf(x.w);
    *(uint2*)(imgB + i) = pk.q;
}

// =====================================================================
// K3: fused bf16 WMMA GEMM + per-row top-5 over this block's N-chunk.
// Block = 8 waves; wave w owns rows [mblock*128 + w*16, +16); its A panel
// (16x512 bf16) lives in 128 VGPRs. All waves share the same N tiles:
// 32-column B tiles are TDM-DMA'd into a double-buffered LDS stage
// (wave 0 issues, TENSORcnt + barrier sync), overlapping DMA of tile t+1
// with compute of tile t. Each k-step runs TWO independent WMMA chains
// (two 16-col sub-tiles) so ds_load latency overlaps matrix issue.
// =====================================================================
__global__ void __launch_bounds__(256)
gemm_topk_kernel(const unsigned short* __restrict__ imgB,
                 const unsigned short* __restrict__ attrB,
                 float* __restrict__ pScore, int* __restrict__ pIdx,
                 int tilesPerChunk)
{
    extern __shared__ char smem[];
    // [0, 2*BUFB): double-buffered 32-row B tile (TDM dest, padded rows)
    float* lscore = (float*)(smem + LISTOFF);
    int*   lidx   = (int*)(smem + LISTOFF + 256 * 8 * TOPK * 4);

    const int tid    = threadIdx.x;
    const int wave   = tid >> 5;
    const int lane   = tid & 31;
    const int l16    = lane & 15;
    const bool hi    = lane >= 16;
    const int mblock = blockIdx.x;
    const int nchunk = blockIdx.y;

    // init per-lane candidate lists (descending, [4] = threshold)
    const int lbase = tid * (8 * TOPK);
    #pragma unroll
    for (int i = 0; i < 8 * TOPK; ++i) { lscore[lbase + i] = NEG_INF; lidx[lbase + i] = 0; }

    // ---- load A: 16 rows x K=512 bf16 in WMMA fragment layout (128 VGPRs) ----
    const int rowA = mblock * 128 + wave * 16 + l16;
    const char* aRow = (const char*)(imgB + (size_t)rowA * FEAT);
    const int offA = hi ? 16 : 0;      // lane<16: K {0-7,16-23}; lane>=16: K {8-15,24-31}
    FragBF a[KSTEPS];
    #pragma unroll
    for (int s = 0; s < KSTEPS; ++s) {
        a[s].q[0] = *(const uint4*)(aRow + 64 * s + offA);
        a[s].q[1] = *(const uint4*)(aRow + 64 * s + offA + 32);
    }

    float thr[8];
    #pragma unroll
    for (int v = 0; v < 8; ++v) thr[v] = NEG_INF;

    const int tile0 = nchunk * tilesPerChunk;
    const int offB  = hi ? 16 : 0;     // lane<16: K {0-7,16-23}; lane>=16: K {8-15,24-31} (bytes)
    const unsigned ldsBase = __builtin_amdgcn_groupstaticsize();  // dynamic-LDS origin

    // prologue: DMA tile 0 into buffer 0
    if (wave == 0)
        tdm_load_tile(attrB + (size_t)tile0 * TROWS * FEAT, ldsBase);

    for (int t = 0; t < tilesPerChunk; ++t) {
        if (wave == 0) {
            if (t + 1 < tilesPerChunk) {
                tdm_load_tile(attrB + (size_t)(tile0 + t + 1) * TROWS * FEAT,
                              ldsBase + (unsigned)(((t + 1) & 1) * BUFB));
                __builtin_amdgcn_s_wait_tensorcnt(1);   // tile t landed, t+1 in flight
            } else {
                __builtin_amdgcn_s_wait_tensorcnt(0);
            }
        }
        __syncthreads();                                // B(t) visible to all waves

        const int col0 = (tile0 + t) * TROWS + l16;     // sub-tile 0 attr index
        const int col1 = col0 + 16;                     // sub-tile 1 attr index
        const char* bRow0 = smem + (t & 1) * BUFB + l16 * BROW + offB;
        const char* bRow1 = bRow0 + 16 * BROW;

        v8f c0 = {0.f, 0.f, 0.f, 0.f, 0.f, 0.f, 0.f, 0.f};
        v8f c1 = {0.f, 0.f, 0.f, 0.f, 0.f, 0.f, 0.f, 0.f};
        #pragma unroll
        for (int s = 0; s < KSTEPS; ++s) {
            FragBF b0, b1;
            b0.q[0] = *(const uint4*)(bRow0 + 64 * s);
            b0.q[1] = *(const uint4*)(bRow0 + 64 * s + 32);
            b1.q[0] = *(const uint4*)(bRow1 + 64 * s);
            b1.q[1] = *(const uint4*)(bRow1 + 64 * s + 32);
            c0 = __builtin_amdgcn_wmma_f32_16x16x32_bf16(
                     false, a[s].v, false, b0.v, (short)0, c0, false, false);
            c1 = __builtin_amdgcn_wmma_f32_16x16x32_bf16(
                     false, a[s].v, false, b1.v, (short)0, c1, false, false);
        }

        // slot v holds score for row (v + 8*hi); two columns per staged tile
        #pragma unroll
        for (int v = 0; v < 8; ++v) {
            float sc = c0[v];
            if (sc > thr[v]) {                       // rare: update sorted LDS list
                float* ls = lscore + lbase + v * TOPK;
                int*   li = lidx   + lbase + v * TOPK;
                float vs = sc; int vi = col0;
                #pragma unroll
                for (int j = 0; j < TOPK; ++j) {
                    if (vs > ls[j]) {
                        float ts = ls[j]; ls[j] = vs; vs = ts;
                        int   ti = li[j]; li[j] = vi; vi = ti;
                    }
                }
                thr[v] = ls[TOPK - 1];
            }
        }
        #pragma unroll
        for (int v = 0; v < 8; ++v) {
            float sc = c1[v];
            if (sc > thr[v]) {
                float* ls = lscore + lbase + v * TOPK;
                int*   li = lidx   + lbase + v * TOPK;
                float vs = sc; int vi = col1;
                #pragma unroll
                for (int j = 0; j < TOPK; ++j) {
                    if (vs > ls[j]) {
                        float ts = ls[j]; ls[j] = vs; vs = ts;
                        int   ti = li[j]; li[j] = vi; vi = ti;
                    }
                }
                thr[v] = ls[TOPK - 1];
            }
        }
        __syncthreads();   // everyone done reading buf(t) before it is re-DMA'd
    }

    __syncthreads();   // all lanes' lists visible

    // ---- wave-local exact merge: row r <- 16 lanes x 5 entries = 80 candidates ----
    for (int r = 0; r < 16; ++r) {
        const int slot = r & 7;
        const int half = (r >= 8) ? 16 : 0;
        float cs[3]; int ci[3];
        #pragma unroll
        for (int u = 0; u < 3; ++u) {
            int cidx = lane + u * 32;
            if (cidx < 16 * TOPK) {
                int srcLane = half + cidx / TOPK;
                int off = (wave * 32 + srcLane) * (8 * TOPK) + slot * TOPK + cidx % TOPK;
                cs[u] = lscore[off]; ci[u] = lidx[off];
            } else { cs[u] = NEG_INF; ci[u] = 0x7FFFFFFF; }
        }
        const int gRow = mblock * 128 + wave * 16 + r;
        #pragma unroll
        for (int k = 0; k < TOPK; ++k) {
            float ms = cs[0]; int mi = ci[0];
            #pragma unroll
            for (int u = 1; u < 3; ++u)
                if (cs[u] > ms || (cs[u] == ms && ci[u] < mi)) { ms = cs[u]; mi = ci[u]; }
            #pragma unroll
            for (int d = 16; d >= 1; d >>= 1) {
                float os = __shfl_xor(ms, d, 32);
                int   oi = __shfl_xor(mi, d, 32);
                if (os > ms || (os == ms && oi < mi)) { ms = os; mi = oi; }
            }
            #pragma unroll
            for (int u = 0; u < 3; ++u)
                if (ci[u] == mi && cs[u] == ms) cs[u] = NEG_INF;
            if (lane == 0) {
                pScore[((size_t)gRow * NCHUNKS + nchunk) * TOPK + k] = ms;
                pIdx  [((size_t)gRow * NCHUNKS + nchunk) * TOPK + k] = mi;
            }
        }
    }
}

// =====================================================================
// K4: final per-row merge of NCHUNKS*5 = 80 chunk-local candidates.
// One wave per image row.
// =====================================================================
__global__ void __launch_bounds__(256)
final_topk_kernel(const float* __restrict__ pScore, const int* __restrict__ pIdx,
                  float* __restrict__ outScore, int* __restrict__ topIdx)
{
    const int wave = threadIdx.x >> 5;
    const int lane = threadIdx.x & 31;
    const int row  = blockIdx.x * 8 + wave;
    const int ncand = NCHUNKS * TOPK;   // 80

    float cs[3]; int ci[3];
    #pragma unroll
    for (int u = 0; u < 3; ++u) {
        int cidx = lane + u * 32;
        if (cidx < ncand) {
            cs[u] = pScore[(size_t)row * ncand + cidx];
            ci[u] = pIdx  [(size_t)row * ncand + cidx];
        } else { cs[u] = NEG_INF; ci[u] = 0x7FFFFFFF; }
    }
    #pragma unroll
    for (int k = 0; k < TOPK; ++k) {
        float ms = cs[0]; int mi = ci[0];
        #pragma unroll
        for (int u = 1; u < 3; ++u)
            if (cs[u] > ms || (cs[u] == ms && ci[u] < mi)) { ms = cs[u]; mi = ci[u]; }
        #pragma unroll
        for (int d = 16; d >= 1; d >>= 1) {
            float os = __shfl_xor(ms, d, 32);
            int   oi = __shfl_xor(mi, d, 32);
            if (os > ms || (os == ms && oi < mi)) { ms = os; mi = oi; }
        }
        #pragma unroll
        for (int u = 0; u < 3; ++u)
            if (ci[u] == mi && cs[u] == ms) cs[u] = NEG_INF;
        if (lane == 0) {
            outScore[(size_t)row * TOPK + k] = ms;
            topIdx [(size_t)row * TOPK + k] = mi;
        }
    }
}

// =====================================================================
// K5: gather selected attr rows, renormalized in full f32 precision.
// One wave per (row, k) pair; 512 floats each.
// =====================================================================
__global__ void __launch_bounds__(256)
gather_kernel(const float* __restrict__ attr, const float* __restrict__ invn,
              const int* __restrict__ topIdx, float* __restrict__ outFeat)
{
    const int wave = threadIdx.x >> 5;
    const int lane = threadIdx.x & 31;
    const int p    = blockIdx.x * 8 + wave;       // (row*5 + k)
    const int idx  = topIdx[p];
    const float sc = invn[idx];
    const float4* src = (const float4*)(attr + (size_t)idx * FEAT);
    float4*       dst = (float4*)(outFeat + (size_t)p * FEAT);
    #pragma unroll
    for (int u = 0; u < 4; ++u) {
        float4 x = src[lane + u * 32];
        x.x *= sc; x.y *= sc; x.z *= sc; x.w *= sc;
        dst[lane + u * 32] = x;
    }
}

// =====================================================================
extern "C" void kernel_launch(void* const* d_in, const int* in_sizes, int n_in,
                              void* d_out, int out_size, void* d_ws, size_t ws_size,
                              hipStream_t stream)
{
    const float* img  = (const float*)d_in[0];   // (batch, 512)
    const float* attr = (const float*)d_in[1];   // (n_attr, 512)
    const int batch  = in_sizes[0] / FEAT;       // 4096
    const int n_attr = in_sizes[1] / FEAT;       // 64000

    float* outFeat  = (float*)d_out;                              // (batch,5,512)
    float* outScore = outFeat + (size_t)batch * TOPK * FEAT;      // (batch,5)

    // workspace carve-up
    char* w = (char*)d_ws;
    unsigned short* attrB = (unsigned short*)w; w += (size_t)n_attr * FEAT * 2;
    unsigned short* imgB  = (unsigned short*)w; w += (size_t)batch  * FEAT * 2;
    float* invn   = (float*)w;  w += (size_t)n_attr * 4;
    int*   topIdx = (int*)w;    w += (size_t)batch * TOPK * 4;
    float* pScore = (float*)w;  w += (size_t)batch * NCHUNKS * TOPK * 4;
    int*   pIdx   = (int*)w;    w += (size_t)batch * NCHUNKS * TOPK * 4;

    const int tilesPerChunk = (n_attr / TROWS) / NCHUNKS;   // 125

    attr_prep_kernel<<<n_attr / 8, 256, 0, stream>>>(attr, attrB, invn);
    img_prep_kernel<<<(batch * FEAT) / (256 * 4), 256, 0, stream>>>(img, imgB);

    dim3 grid(batch / 128, NCHUNKS);
    gemm_topk_kernel<<<grid, 256, LDS_SZ, stream>>>(imgB, attrB, pScore, pIdx,
                                                    tilesPerChunk);

    final_topk_kernel<<<batch / 8, 256, 0, stream>>>(pScore, pIdx, outScore, topIdx);
    gather_kernel<<<(batch * TOPK) / 8, 256, 0, stream>>>(attr, invn, topIdx, outFeat);
}
